// MyNet_14894946583033
// MI455X (gfx1250) — compile-verified, matrix operand order checked by reading, not run on hardware
//
#include <hip/hip_runtime.h>
#include <hip/hip_bf16.h>

// ---------------- problem constants ----------------
#define DETN   648
#define NVIEW  360
#define NIMG   512
#define DPAD   672            // detector dim padded to 21*32 for WMMA K / 42*16 for N
#define KPAD   160            // im2col K = 144 padded to 5*32
#define DUC    0.48f          // DET_SIZE*SID/SDD = 0.8*750/1250
#define SIDC   750.0f
#define PIXC   0.5f

typedef __attribute__((ext_vector_type(16))) _Float16 v16h;
typedef __attribute__((ext_vector_type(8)))  _Float16 v8h;
typedef __attribute__((ext_vector_type(8)))  float    v8f;

union AV { v16h v; v8h h[2]; };

// ---------------- H^T build: HT[j][i] = du * h(j-i), zero-padded to 672 ----
__global__ void build_ht_k(_Float16* HT) {
    int t = blockIdx.x * blockDim.x + threadIdx.x;
    if (t >= DPAD * DPAD) return;
    int j = t / DPAD, i = t % DPAD;
    float val = 0.0f;
    if (i < DETN && j < DETN) {
        int n = j - i;
        if (n == 0) {
            val = 1.0f / (8.0f * DUC * DUC);
        } else if (n & 1) {
            float pn = 3.14159265358979323846f * (float)n;
            val = -1.0f / (2.0f * pn * pn * DUC * DUC);
        }
        val *= DUC;
    }
    HT[t] = (_Float16)val;
}

// ---------------- pack conv weights [16][16][3][3] -> WT[o][k=tap*16+c] f16 --
__global__ void pack_wt_k(const float* __restrict__ W, _Float16* __restrict__ WT) {
    int t = blockIdx.x * blockDim.x + threadIdx.x;
    if (t >= 16 * KPAD) return;
    int o = t / KPAD, k = t % KPAD;
    float v = 0.0f;
    if (k < 144) {
        int tap = k >> 4, c = k & 15;
        v = W[(o * 16 + c) * 9 + tap];
    }
    WT[t] = (_Float16)v;
}

// ---------------- conv1: 1->16 direct, NCHW f32 in -> NHWC f16 out ----------
__global__ void conv1_k(const float* __restrict__ x, const float* __restrict__ W1,
                        const float* __restrict__ b1, _Float16* __restrict__ t1) {
    int t = blockIdx.x * blockDim.x + threadIdx.x;
    if (t >= 2 * NVIEW * DETN) return;
    int d = t % DETN;
    int v = (t / DETN) % NVIEW;
    int b = t / (DETN * NVIEW);
    float vin[9];
    #pragma unroll
    for (int dy = 0; dy < 3; ++dy)
        #pragma unroll
        for (int dx = 0; dx < 3; ++dx) {
            int yy = v + dy - 1, xx = d + dx - 1;
            vin[dy * 3 + dx] = (yy >= 0 && yy < NVIEW && xx >= 0 && xx < DETN)
                               ? x[(b * NVIEW + yy) * DETN + xx] : 0.0f;
        }
    _Float16* o = t1 + (size_t)t * 16;
    #pragma unroll
    for (int c = 0; c < 16; ++c) {
        float a = b1[c];
        #pragma unroll
        for (int k = 0; k < 9; ++k) a += W1[c * 9 + k] * vin[k];
        o[c] = (_Float16)a;
    }
}

// ------------- WMMA 3x3x16->16 conv tile core (one wave, 16 pixels) ---------
// f16 NHWC input; patch staged via async global->LDS copies (ASYNCcnt path),
// with pre-zeroed LDS providing the SAME-padding halo.
__device__ inline v8f conv_tile_wmma(const _Float16* __restrict__ inb, int H, int W,
                                     int y, int x0, const _Float16* __restrict__ WT,
                                     _Float16* patch) {
    int lane = threadIdx.x & 31;

    // zero the 3x18x16 f16 patch (216 qwords) so OOB taps read 0
    unsigned long long* pz = (unsigned long long*)patch;
    #pragma unroll
    for (int e = 0; e < 7; ++e) {
        int idx = lane + e * 32;
        if (idx < 216) pz[idx] = 0ull;
    }
    __syncthreads();

    // async copy valid pixels: 54 pixels x 2 x 16B chunks, global -> LDS
    unsigned ldsbase = (unsigned)(uintptr_t)patch;
    for (int e = lane; e < 108; e += 32) {
        int hc = e & 1;
        int p  = e >> 1;                 // p = yl*18 + xl
        int xl = p % 18, yl = p / 18;
        int gy = y + yl - 1, gx = x0 + xl - 1;
        if (gy >= 0 && gy < H && gx >= 0 && gx < W) {
            const _Float16* src = inb + ((size_t)(gy * W + gx)) * 16 + hc * 8;
            unsigned dst = ldsbase + (unsigned)(p * 32 + hc * 16);
            asm volatile("global_load_async_to_lds_b128 %0, %1, off"
                         :: "v"(dst), "v"(src) : "memory");
        }
    }
    asm volatile("s_wait_asynccnt 0x0" ::: "memory");
    __syncthreads();

    int sel = lane >> 4;            // wave half selects k chunk offset
    int m   = lane & 15;            // A row = pixel index / B row = out channel
    v8f acc = {};
    #pragma unroll
    for (int ks = 0; ks < 5; ++ks) {
        int k0 = ks * 32 + sel * 8;
        int k1 = k0 + 16;
        AV a, bb;
        {   int tap = k0 >> 4, ch = k0 & 15;
            int dy = tap / 3, dx = tap % 3;
            a.h[0] = *(const v8h*)(patch + ((dy * 18 + m + dx) * 16 + ch)); }
        if (k1 < 144) {
            int tap = k1 >> 4, ch = k1 & 15;
            int dy = tap / 3, dx = tap % 3;
            a.h[1] = *(const v8h*)(patch + ((dy * 18 + m + dx) * 16 + ch));
        } else {
            v8h z = {}; a.h[1] = z;             // k >= 144 zero-padded
        }
        bb.h[0] = *(const v8h*)(WT + m * KPAD + k0);
        bb.h[1] = *(const v8h*)(WT + m * KPAD + k1);
        acc = __builtin_amdgcn_wmma_f32_16x16x32_f16(false, a.v, false, bb.v,
                                                     (short)0, acc, false, false);
    }
    return acc;
}

// ---- conv2: 16->16 WMMA; fuse bias + cos-weight + f16; scatter into GEMM A -
// A layout: [b][v][c][DPAD] f16  (row = (b*360+v)*16 + c, K = detector)
__global__ void conv2_k(const _Float16* __restrict__ t1, const _Float16* __restrict__ WT2,
                        const float* __restrict__ b2, _Float16* __restrict__ A) {
    __shared__ _Float16 patch[3 * 18 * 16];
    int blk  = blockIdx.x;
    int tile = blk % 41;
    int v    = (blk / 41) % NVIEW;
    int b    = blk / (41 * NVIEW);
    int x0   = tile * 16;
    const _Float16* inb = t1 + (size_t)b * NVIEW * DETN * 16;
    v8f acc = conv_tile_wmma(inb, NVIEW, DETN, v, x0, WT2, patch);

    int lane = threadIdx.x & 31;
    int ch   = lane & 15;               // C: N = out channel
    int sel  = lane >> 4;
    float bias = b2[ch];
    size_t rowbase = ((size_t)(b * NVIEW + v) * 16 + ch) * DPAD;
    #pragma unroll
    for (int r = 0; r < 8; ++r) {
        int d = x0 + r + 8 * sel;       // C: M = pixel
        if (d < DETN) {
            float u  = ((float)d - 323.5f) * DUC;
            float cw = SIDC * rsqrtf(SIDC * SIDC + u * u);
            A[rowbase + d] = (_Float16)((acc[r] + bias) * cw);
        }
    }
}

// ---- ramp-filter GEMM: Filt = A(11520x672,f16) * H(672x672) via WMMA -------
// B operand loaded A-style from HT (row-major H^T). Output channel-fastest:
// Filt[b][v][d][c] f32 for coalesced backprojection gathers.
__global__ void fgemm_k(const _Float16* __restrict__ A, const _Float16* __restrict__ HT,
                        float* __restrict__ Filt) {
    int blk = blockIdx.x;
    int dt  = blk % 42;                 // N tile (detector out)
    int bv  = blk / 42;                 // 0..719  (b*360+v)
    int d0  = dt * 16;
    int lane = threadIdx.x & 31;
    int m    = lane & 15;
    int sel  = lane >> 4;
    const v8h* arow = (const v8h*)(A  + ((size_t)bv * 16 + m) * DPAD);
    const v8h* brow = (const v8h*)(HT + (size_t)(d0 + m) * DPAD);
    v8f acc = {};
    #pragma unroll 3
    for (int ks = 0; ks < 21; ++ks) {
        int c0 = ks * 4 + sel;          // v8h chunk index: (ks*32 + sel*8)/8
        AV a, bb;
        a.h[0]  = arow[c0];  a.h[1]  = arow[c0 + 2];
        bb.h[0] = brow[c0];  bb.h[1] = brow[c0 + 2];
        acc = __builtin_amdgcn_wmma_f32_16x16x32_f16(false, a.v, false, bb.v,
                                                     (short)0, acc, false, false);
    }
    // C: N = m -> detector d0+m, M = r+8*sel -> channel. Store [bv][d][c].
    size_t base = ((size_t)bv * DPAD + d0 + m) * 16;
    #pragma unroll
    for (int r = 0; r < 8; ++r) Filt[base + r + 8 * sel] = acc[r];
}

// ---------------- backprojection: gather, 16 channels vectorized ------------
// writes img as f16 NHWC (the following WMMA conv quantizes to f16 anyway)
__global__ void backproj_k(const float* __restrict__ Filt, _Float16* __restrict__ img) {
    int t = blockIdx.x * blockDim.x + threadIdx.x;
    if (t >= 2 * NIMG * NIMG) return;
    int x = t % NIMG;
    int y = (t / NIMG) % NIMG;
    int b = t / (NIMG * NIMG);
    float X = ((float)x - 255.5f) * PIXC;
    float Y = ((float)y - 255.5f) * PIXC;
    float acc[16];
    #pragma unroll
    for (int c = 0; c < 16; ++c) acc[c] = 0.0f;
    const float* Fb = Filt + (size_t)b * NVIEW * DPAD * 16;
    for (int v = 0; v < NVIEW; ++v) {
        float beta = (float)v * (3.14159265358979323846f / 180.0f);
        float sb, cb;
        __sincosf(beta, &sb, &cb);
        float L    = SIDC - (X * cb + Y * sb);
        float invL = 1.0f / L;
        float up   = SIDC * (-X * sb + Y * cb) * invL;
        float w    = SIDC * invL; w = w * w;
        float idx  = up * (1.0f / DUC) + 323.5f;
        float fi   = floorf(idx);
        int   i0   = (int)fi;
        float frac = idx - fi;
        float valid = (i0 >= 0 && i0 < DETN - 1) ? 1.0f : 0.0f;
        int   i0c  = min(max(i0, 0), DETN - 2);
        const float4* p4 = (const float4*)(Fb + ((size_t)v * DPAD + i0c) * 16);
        // speculative prefetch of the next view's neighborhood (global_prefetch_b8)
        if (v + 1 < NVIEW)
            __builtin_prefetch(Fb + ((size_t)(v + 1) * DPAD + i0c) * 16, 0, 0);
        float4 a0 = p4[0], a1 = p4[1], a2 = p4[2], a3 = p4[3];
        float4 c0 = p4[4], c1 = p4[5], c2 = p4[6], c3 = p4[7];
        float wv = w * valid;
        float f1 = frac * wv, f0 = (1.0f - frac) * wv;
        acc[0]  += a0.x * f0 + c0.x * f1;  acc[1]  += a0.y * f0 + c0.y * f1;
        acc[2]  += a0.z * f0 + c0.z * f1;  acc[3]  += a0.w * f0 + c0.w * f1;
        acc[4]  += a1.x * f0 + c1.x * f1;  acc[5]  += a1.y * f0 + c1.y * f1;
        acc[6]  += a1.z * f0 + c1.z * f1;  acc[7]  += a1.w * f0 + c1.w * f1;
        acc[8]  += a2.x * f0 + c2.x * f1;  acc[9]  += a2.y * f0 + c2.y * f1;
        acc[10] += a2.z * f0 + c2.z * f1;  acc[11] += a2.w * f0 + c2.w * f1;
        acc[12] += a3.x * f0 + c3.x * f1;  acc[13] += a3.y * f0 + c3.y * f1;
        acc[14] += a3.z * f0 + c3.z * f1;  acc[15] += a3.w * f0 + c3.w * f1;
    }
    const float dbeta = 6.283185307179586f / 360.0f;
    _Float16* o = img + (size_t)t * 16;
    #pragma unroll
    for (int c = 0; c < 16; ++c) o[c] = (_Float16)(acc[c] * dbeta);
}

// ---------------- conv3: 16->16 WMMA on image, NHWC f32 out -----------------
__global__ void conv3_k(const _Float16* __restrict__ img, const _Float16* __restrict__ WT3,
                        const float* __restrict__ b3, float* __restrict__ h3) {
    __shared__ _Float16 patch[3 * 18 * 16];
    int blk  = blockIdx.x;
    int tile = blk % 32;
    int y    = (blk / 32) % NIMG;
    int b    = blk / (32 * NIMG);
    int x0   = tile * 16;
    const _Float16* inb = img + (size_t)b * NIMG * NIMG * 16;
    v8f acc = conv_tile_wmma(inb, NIMG, NIMG, y, x0, WT3, patch);

    int lane = threadIdx.x & 31;
    int ch   = lane & 15;
    int sel  = lane >> 4;
    float bias = b3[ch];
    float* ob = h3 + (((size_t)(b * NIMG + y) * NIMG) + x0) * 16;
    #pragma unroll
    for (int r = 0; r < 8; ++r) {
        int m = r + 8 * sel;
        ob[(size_t)m * 16 + ch] = acc[r] + bias;
    }
}

// ---------------- conv4: 16->1 direct, NHWC in -> NCHW out ------------------
__global__ void conv4_k(const float* __restrict__ h3, const float* __restrict__ W4,
                        const float* __restrict__ b4, float* __restrict__ out) {
    int t = blockIdx.x * blockDim.x + threadIdx.x;
    if (t >= 2 * NIMG * NIMG) return;
    int x = t % NIMG;
    int y = (t / NIMG) % NIMG;
    int b = t / (NIMG * NIMG);
    float a = b4[0];
    #pragma unroll
    for (int dy = 0; dy < 3; ++dy) {
        int yy = y + dy - 1;
        if (yy < 0 || yy >= NIMG) continue;
        #pragma unroll
        for (int dx = 0; dx < 3; ++dx) {
            int xx = x + dx - 1;
            if (xx < 0 || xx >= NIMG) continue;
            const float* p = h3 + ((size_t)(b * NIMG + yy) * NIMG + xx) * 16;
            int tap = dy * 3 + dx;
            #pragma unroll
            for (int c = 0; c < 16; ++c) a += p[c] * W4[c * 9 + tap];
        }
    }
    out[t] = a;
}

// ---------------- workspace layout (liveness-overlapped, ~90 MB peak) -------
static constexpr size_t OFF_T1   = 0;                       // 14,929,920 B  f16 NHWC sino16
static constexpr size_t OFF_A    = 33554432;                // 15,482,880 B  f16 GEMM A
static constexpr size_t A_BYTES  = (size_t)2 * NVIEW * 16 * DPAD * 2;
static constexpr size_t OFF_HT   = 50331648;                //    903,168 B  f16 H^T
static constexpr size_t OFF_FILT = 58720256;                // 30,965,760 B  f32 filtered
static constexpr size_t OFF_IMG  = 0;                       // 16,777,216 B  f16 (reuses T1)
static constexpr size_t OFF_H3   = 33554432;                // 33,554,432 B  f32 (reuses A/HT/Filt-head)
static constexpr size_t OFF_WT2  = 89784320;                //      5,120 B
static constexpr size_t OFF_WT3  = 89800704;                //      5,120 B

extern "C" void kernel_launch(void* const* d_in, const int* in_sizes, int n_in,
                              void* d_out, int out_size, void* d_ws, size_t ws_size,
                              hipStream_t stream) {
    const float* x  = (const float*)d_in[0];
    const float* W1 = (const float*)d_in[1];
    const float* b1 = (const float*)d_in[2];
    const float* W2 = (const float*)d_in[3];
    const float* b2 = (const float*)d_in[4];
    const float* W3 = (const float*)d_in[5];
    const float* b3 = (const float*)d_in[6];
    const float* W4 = (const float*)d_in[7];
    const float* b4 = (const float*)d_in[8];

    char* ws = (char*)d_ws;
    _Float16*  t1   = (_Float16*)(ws + OFF_T1);
    _Float16*  Af   = (_Float16*)(ws + OFF_A);
    _Float16*  HT   = (_Float16*)(ws + OFF_HT);
    float*     Filt = (float*)(ws + OFF_FILT);
    _Float16*  img  = (_Float16*)(ws + OFF_IMG);
    float*     h3   = (float*)(ws + OFF_H3);
    _Float16*  WT2  = (_Float16*)(ws + OFF_WT2);
    _Float16*  WT3  = (_Float16*)(ws + OFF_WT3);

    build_ht_k<<<(DPAD * DPAD + 255) / 256, 256, 0, stream>>>(HT);
    pack_wt_k<<<(16 * KPAD + 255) / 256, 256, 0, stream>>>(W2, WT2);
    pack_wt_k<<<(16 * KPAD + 255) / 256, 256, 0, stream>>>(W3, WT3);
    conv1_k<<<(2 * NVIEW * DETN + 255) / 256, 256, 0, stream>>>(x, W1, b1, t1);
    hipMemsetAsync(Af, 0, A_BYTES, stream);   // zero K/N padding columns of A
    conv2_k<<<2 * NVIEW * 41, 32, 0, stream>>>(t1, WT2, b2, Af);
    fgemm_k<<<2 * NVIEW * 42, 32, 0, stream>>>(Af, HT, Filt);
    backproj_k<<<(2 * NIMG * NIMG + 255) / 256, 256, 0, stream>>>(Filt, img);
    conv3_k<<<2 * NIMG * 32, 32, 0, stream>>>(img, WT3, b3, h3);
    conv4_k<<<(2 * NIMG * NIMG + 255) / 256, 256, 0, stream>>>(h3, W4, b4, (float*)d_out);
}
